// HybridEdgeRegressor_36197984370763
// MI455X (gfx1250) — matrix operand order repaired and checked by reading, not compiled, not run on hardware
//
#include <hip/hip_runtime.h>

// ---------------------------------------------------------------------------
// HybridEdgeRegressor for MI455X (gfx1250, wave32, WMMA bf16 16x16x32).
//
// Input slot map (setup_inputs() dict insertion order, depth-first):
//   0: x [N,128] f32     1: edge_index [2,E] i32
//   2+9i (i=0..2) GAT layer i: +0 lin_l.W  +1 lin_l.b  +2 lin_r.W  +3 lin_r.b
//                              +4 att      +5 Wres     +6 bias     +7 ln_g  +8 ln_b
//   29..32: local  l1.W l1.b l2.W l2.b
//   33..36: edge_init l1.W l1.b l2.W l2.b
//   37+8j (j=0..2) upd j: +0 l1.W +1 l1.b +2 ln1_g +3 ln1_b +4 l2.W +5 l2.b +6 ln2_g +7 ln2_b
//   61..64: edge_out l1.W l1.b l2.W l2.b
// ---------------------------------------------------------------------------

#define NN      10000
#define NE      160000
#define DIN     128
#define HDIM    256
#define LCDIM   64
#define EHDIM   256
#define XCD     320
#define EIK     1280
#define LDK     40   // u16 stride: 80B rows -> 16B aligned, conflict-free frag reads

typedef __attribute__((ext_vector_type(16))) __bf16 v16bf;
typedef __attribute__((ext_vector_type(8)))  float  v8f;

// ------------------------- device helpers ----------------------------------
__device__ __forceinline__ unsigned short f32_bf16_rne(float f) {
  union { float f; unsigned u; } x; x.f = f;
  unsigned r = x.u + 0x7FFFu + ((x.u >> 16) & 1u);
  return (unsigned short)(r >> 16);
}
__device__ __forceinline__ unsigned pack2bf(float a, float b) {
  return (unsigned)f32_bf16_rne(a) | ((unsigned)f32_bf16_rne(b) << 16);
}
__device__ __forceinline__ unsigned ord_enc(float f) {
  unsigned u = __float_as_uint(f);
  return (u & 0x80000000u) ? ~u : (u | 0x80000000u);
}
__device__ __forceinline__ float ord_dec(unsigned u) {
  return __uint_as_float((u & 0x80000000u) ? (u ^ 0x80000000u) : ~u);
}

// A fragment: 16x32 bf16, ISA 7.12.2 layout. sm is [16][LDK] (row, k).
__device__ __forceinline__ v16bf frag_ld_a(const unsigned short* sm, int lane) {
  int r  = lane & 15;
  int kb = (lane & 16) ? 8 : 0;
  union { unsigned short u[16]; v16bf v; } f;
#pragma unroll
  for (int vv = 0; vv < 8; ++vv) {
    int k = (vv < 4) ? (kb + 2 * vv) : (16 + kb + 2 * (vv - 4));
    f.u[2 * vv]     = sm[r * LDK + k];
    f.u[2 * vv + 1] = sm[r * LDK + k + 1];
  }
  return f.v;
}
// B fragment: 32x16 bf16 (K x N); sm is [16][LDK] indexed (col, k).
__device__ __forceinline__ v16bf frag_ld_b(const unsigned short* sm, int lane) {
  int c    = lane & 15;
  int koff = (lane & 16) ? 8 : 0;
  union { unsigned short u[16]; v16bf v; } f;
#pragma unroll
  for (int vv = 0; vv < 8; ++vv) {
    int k = (vv < 4) ? (koff + 2 * vv) : (16 + koff + 2 * (vv - 4));
    f.u[2 * vv]     = sm[c * LDK + k];
    f.u[2 * vv + 1] = sm[c * LDK + k + 1];
  }
  return f.v;
}

// ------------------------- generic bf16 WMMA GEMM --------------------------
// C[M,Nn] = act(A_f32[M,K] @ (WT_bf16[Nn,K])^T + bias) ; 4 waves, 64x64 tile.
// WT is the transposed weight (row n holds K contiguous bf16).
__global__ __launch_bounds__(128) void k_gemm(
    const float* __restrict__ A, const unsigned short* __restrict__ WT,
    const float* __restrict__ bias, float* __restrict__ C,
    int M, int K, int Nn, int do_relu)
{
  __shared__ unsigned short As[64 * LDK];
  __shared__ unsigned short Bs[64 * LDK];
  const int tid = threadIdx.x, lane = tid & 31, wave = tid >> 5;
  const int rowBlk = blockIdx.x * 64, colBlk = blockIdx.y * 64;

  v8f acc[4];
#pragma unroll
  for (int t = 0; t < 4; ++t)
#pragma unroll
    for (int r = 0; r < 8; ++r) acc[t][r] = 0.0f;

  for (int k0 = 0; k0 < K; k0 += 32) {
    // A tile 64x32: float4 load -> 4x bf16 -> one b64 LDS store per chunk
#pragma unroll
    for (int i = 0; i < 4; ++i) {
      int q = tid + i * 128;                 // 0..511
      int r = q >> 3, kk = (q & 7) * 4;
      int gr = rowBlk + r; if (gr >= M) gr = M - 1;     // branchless tail clamp
      float4 f4 = *(const float4*)(A + (size_t)((unsigned)gr * (unsigned)K + (unsigned)(k0 + kk)));
      uint2 p; p.x = pack2bf(f4.x, f4.y); p.y = pack2bf(f4.z, f4.w);
      *(uint2*)&As[r * LDK + kk] = p;
    }
    // B tile: 8 contiguous bf16 along K per chunk -> b128 LDS store
#pragma unroll
    for (int i = 0; i < 2; ++i) {
      int q = tid + i * 128;                 // 0..255
      int n = q >> 2, kk = (q & 3) * 8;
      uint4 w = *(const uint4*)(WT + (size_t)((unsigned)(colBlk + n) * (unsigned)K + (unsigned)(k0 + kk)));
      *(uint4*)&Bs[n * LDK + kk] = w;
    }
    __syncthreads();
    v16bf a = frag_ld_a(&As[(wave * 16) * LDK], lane);
#pragma unroll
    for (int t = 0; t < 4; ++t) {
      v16bf b = frag_ld_b(&Bs[(t * 16) * LDK], lane);
      acc[t] = __builtin_amdgcn_wmma_f32_16x16x32_bf16(
          false, a, false, b, (short)0, acc[t], false, false);
    }
    __syncthreads();
  }
  const int mloc = wave * 16 + ((lane >> 4) << 3);
  const int nloc = lane & 15;
#pragma unroll
  for (int t = 0; t < 4; ++t) {
    int gn = colBlk + t * 16 + nloc;
    float bv = bias ? bias[gn] : 0.0f;
#pragma unroll
    for (int r = 0; r < 8; ++r) {
      int gm = rowBlk + mloc + r;
      if (gm < M) {
        float v = acc[t][r] + bv;
        if (do_relu) v = fmaxf(v, 0.0f);
        C[(size_t)((unsigned)gm * (unsigned)Nn + (unsigned)gn)] = v;
      }
    }
  }
}

// edge_init GEMM: A row e = [xs, xd, xs-xd, xs*xd] over xc[*,320]; K=1280, Nn=256.
__global__ __launch_bounds__(128) void k_gemm_edgefeat(
    const float* __restrict__ xc, const int* __restrict__ src, const int* __restrict__ dst,
    const unsigned short* __restrict__ WT, const float* __restrict__ bias,
    float* __restrict__ C, int M, int do_relu)
{
  __shared__ unsigned short As[64 * LDK];
  __shared__ unsigned short Bs[64 * LDK];
  __shared__ int ss[64], dd[64];
  const int tid = threadIdx.x, lane = tid & 31, wave = tid >> 5;
  const int rowBlk = blockIdx.x * 64, colBlk = blockIdx.y * 64;
  if (tid < 64) {
    int ge = rowBlk + tid; if (ge >= M) ge = M - 1;
    ss[tid] = src[ge];
    dd[tid] = dst[ge];
  }
  __syncthreads();

  v8f acc[4];
#pragma unroll
  for (int t = 0; t < 4; ++t)
#pragma unroll
    for (int r = 0; r < 8; ++r) acc[t][r] = 0.0f;

  for (int k0 = 0; k0 < EIK; k0 += 32) {
    const int seg = k0 / 320;              // 320 % 32 == 0 -> uniform per step
    const int fb  = k0 - seg * 320;
#pragma unroll
    for (int i = 0; i < 4; ++i) {
      int q = tid + i * 128;
      int r = q >> 3, kk = (q & 7) * 4;
      int f = fb + kk;
      float4 a4 = *(const float4*)(xc + (size_t)((unsigned)ss[r] * XCD + (unsigned)f));
      float4 b4 = *(const float4*)(xc + (size_t)((unsigned)dd[r] * XCD + (unsigned)f));
      float o0, o1, o2, o3;
      if (seg == 0)      { o0 = a4.x; o1 = a4.y; o2 = a4.z; o3 = a4.w; }
      else if (seg == 1) { o0 = b4.x; o1 = b4.y; o2 = b4.z; o3 = b4.w; }
      else if (seg == 2) { o0 = a4.x - b4.x; o1 = a4.y - b4.y; o2 = a4.z - b4.z; o3 = a4.w - b4.w; }
      else               { o0 = a4.x * b4.x; o1 = a4.y * b4.y; o2 = a4.z * b4.z; o3 = a4.w * b4.w; }
      uint2 p; p.x = pack2bf(o0, o1); p.y = pack2bf(o2, o3);
      *(uint2*)&As[r * LDK + kk] = p;
    }
#pragma unroll
    for (int i = 0; i < 2; ++i) {
      int q = tid + i * 128;
      int n = q >> 2, kk = (q & 3) * 8;
      uint4 w = *(const uint4*)(WT + (size_t)((unsigned)(colBlk + n) * (unsigned)EIK + (unsigned)(k0 + kk)));
      *(uint4*)&Bs[n * LDK + kk] = w;
    }
    __syncthreads();
    v16bf a = frag_ld_a(&As[(wave * 16) * LDK], lane);
#pragma unroll
    for (int t = 0; t < 4; ++t) {
      v16bf b = frag_ld_b(&Bs[(t * 16) * LDK], lane);
      acc[t] = __builtin_amdgcn_wmma_f32_16x16x32_bf16(
          false, a, false, b, (short)0, acc[t], false, false);
    }
    __syncthreads();
  }
  const int mloc = wave * 16 + ((lane >> 4) << 3);
  const int nloc = lane & 15;
#pragma unroll
  for (int t = 0; t < 4; ++t) {
    int gn = colBlk + t * 16 + nloc;
    float bv = bias[gn];
#pragma unroll
    for (int r = 0; r < 8; ++r) {
      int gm = rowBlk + mloc + r;
      if (gm < M) {
        float v = acc[t][r] + bv;
        if (do_relu) v = fmaxf(v, 0.0f);
        C[(size_t)((unsigned)gm * EHDIM + (unsigned)gn)] = v;
      }
    }
  }
}

// ---------------- LayerNorm over 256-wide rows (wave per row) --------------
// out = [resid +] act( LN(inp [+ add1] [+ addb]) * g + b )
__global__ __launch_bounds__(256) void k_layernorm(
    const float* __restrict__ inp, const float* __restrict__ add1,
    const float* __restrict__ addb, const float* __restrict__ g,
    const float* __restrict__ b, const float* __restrict__ resid,
    float* __restrict__ out, int rows, int relu_out)
{
  int row = blockIdx.x * 8 + (threadIdx.x >> 5);
  if (row >= rows) return;
  int lane = threadIdx.x & 31;
  size_t base = (size_t)((unsigned)row) * 256u;
  float v[8], s = 0.0f;
#pragma unroll
  for (int h = 0; h < 2; ++h) {
    int c = h * 128 + lane * 4;
    float4 x4 = *(const float4*)(inp + base + c);
    if (add1) {
      float4 a4 = *(const float4*)(add1 + base + c);
      x4.x += a4.x; x4.y += a4.y; x4.z += a4.z; x4.w += a4.w;
    }
    if (addb) {
      float4 b4 = *(const float4*)(addb + c);
      x4.x += b4.x; x4.y += b4.y; x4.z += b4.z; x4.w += b4.w;
    }
    v[h * 4 + 0] = x4.x; v[h * 4 + 1] = x4.y; v[h * 4 + 2] = x4.z; v[h * 4 + 3] = x4.w;
    s += x4.x + x4.y + x4.z + x4.w;
  }
#pragma unroll
  for (int o = 16; o > 0; o >>= 1) s += __shfl_xor(s, o, 32);
  float mean = s * (1.0f / 256.0f), var = 0.0f;
#pragma unroll
  for (int i = 0; i < 8; ++i) { float d = v[i] - mean; var += d * d; }
#pragma unroll
  for (int o = 16; o > 0; o >>= 1) var += __shfl_xor(var, o, 32);
  float rstd = rsqrtf(var * (1.0f / 256.0f) + 1e-5f);
#pragma unroll
  for (int h = 0; h < 2; ++h) {
    int c = h * 128 + lane * 4;
    float4 g4 = *(const float4*)(g + c);
    float4 b4 = *(const float4*)(b + c);
    float4 y;
    y.x = (v[h * 4 + 0] - mean) * rstd * g4.x + b4.x;
    y.y = (v[h * 4 + 1] - mean) * rstd * g4.y + b4.y;
    y.z = (v[h * 4 + 2] - mean) * rstd * g4.z + b4.z;
    y.w = (v[h * 4 + 3] - mean) * rstd * g4.w + b4.w;
    if (relu_out) {
      y.x = fmaxf(y.x, 0.f); y.y = fmaxf(y.y, 0.f);
      y.z = fmaxf(y.z, 0.f); y.w = fmaxf(y.w, 0.f);
    }
    if (resid) {
      float4 r4 = *(const float4*)(resid + base + c);
      y.x += r4.x; y.y += r4.y; y.z += r4.z; y.w += r4.w;
    }
    *(float4*)(out + base + c) = y;
  }
}

// ---------------- GATv2 attention (E real edges + N self loops) ------------
__global__ __launch_bounds__(256) void k_att_logit(
    const float* __restrict__ xl, const float* __restrict__ xr,
    const int* __restrict__ src, const int* __restrict__ dst,
    const float* __restrict__ att, float* __restrict__ logit,
    unsigned* __restrict__ lmax, int E, int Nv)
{
  int e = blockIdx.x * 8 + (threadIdx.x >> 5);
  if (e >= E + Nv) return;
  int lane = threadIdx.x & 31;
  int s = (e < E) ? src[e] : (e - E);
  int d = (e < E) ? dst[e] : (e - E);
  int h = lane >> 3;
  int c0 = lane * 8;                               // 8 channels, one head per lane
  const float* pl = xl + (size_t)((unsigned)s * 256u + (unsigned)c0);
  const float* pr = xr + (size_t)((unsigned)d * 256u + (unsigned)c0);
  float p = 0.0f;
#pragma unroll
  for (int q = 0; q < 2; ++q) {
    float4 l4 = *(const float4*)(pl + q * 4);
    float4 r4 = *(const float4*)(pr + q * 4);
    float4 a4 = *(const float4*)(att + c0 + q * 4);
    float t0 = l4.x + r4.x, t1 = l4.y + r4.y, t2 = l4.z + r4.z, t3 = l4.w + r4.w;
    t0 = (t0 > 0.f) ? t0 : 0.2f * t0;  t1 = (t1 > 0.f) ? t1 : 0.2f * t1;
    t2 = (t2 > 0.f) ? t2 : 0.2f * t2;  t3 = (t3 > 0.f) ? t3 : 0.2f * t3;
    p += t0 * a4.x + t1 * a4.y + t2 * a4.z + t3 * a4.w;
  }
#pragma unroll
  for (int o = 1; o < 8; o <<= 1) p += __shfl_xor(p, o, 32);
  if ((lane & 7) == 0) {
    logit[(size_t)((unsigned)e * 4u + (unsigned)h)] = p;
    atomicMax(&lmax[d * 4 + h], ord_enc(p));
  }
}

__global__ __launch_bounds__(256) void k_att_ex(
    float* __restrict__ logit_ex, const unsigned* __restrict__ lmax,
    float* __restrict__ den, const int* __restrict__ dst, int E, int Nv)
{
  int idx = blockIdx.x * 256 + threadIdx.x;
  int e = idx >> 2, h = idx & 3;
  if (e >= E + Nv) return;
  int d = (e < E) ? dst[e] : (e - E);
  float ex = __expf(logit_ex[idx] - ord_dec(lmax[d * 4 + h]));
  logit_ex[idx] = ex;
  atomicAdd(&den[d * 4 + h], ex);
}

__global__ __launch_bounds__(256) void k_att_scatter(
    const float* __restrict__ xl, const float* __restrict__ ex,
    const float* __restrict__ den, const int* __restrict__ src,
    const int* __restrict__ dst, float* __restrict__ outp, int E, int Nv)
{
  int e = blockIdx.x * 8 + (threadIdx.x >> 5);
  if (e >= E + Nv) return;
  int lane = threadIdx.x & 31;
  int s = (e < E) ? src[e] : (e - E);
  int d = (e < E) ? dst[e] : (e - E);
  int h = lane >> 3;
  int c0 = lane * 8;
  float alpha = ex[(size_t)((unsigned)e * 4u + (unsigned)h)] / den[d * 4 + h];
  const float* pl = xl + (size_t)((unsigned)s * 256u + (unsigned)c0);
  float* po = outp + (size_t)((unsigned)d * 256u + (unsigned)c0);
#pragma unroll
  for (int q = 0; q < 2; ++q) {
    float4 l4 = *(const float4*)(pl + q * 4);
    atomicAdd(po + q * 4 + 0, alpha * l4.x);
    atomicAdd(po + q * 4 + 1, alpha * l4.y);
    atomicAdd(po + q * 4 + 2, alpha * l4.z);
    atomicAdd(po + q * 4 + 3, alpha * l4.w);
  }
}

// ---------------- local context gather + misc elementwise ------------------
__global__ __launch_bounds__(256) void k_ctx_scatter(
    const float* __restrict__ x, const int* __restrict__ src,
    const int* __restrict__ dst, float* __restrict__ ctx,
    float* __restrict__ cnt, int E)
{
  int e = blockIdx.x * 8 + (threadIdx.x >> 5);
  if (e >= E) return;
  int lane = threadIdx.x & 31;
  int s = src[e], d = dst[e];
  int c0 = lane * 8;
  const float* px = x + (size_t)((unsigned)d * 256u + (unsigned)c0);
  float* pc = ctx + (size_t)((unsigned)s * 256u + (unsigned)c0);
#pragma unroll
  for (int q = 0; q < 2; ++q) {
    float4 x4 = *(const float4*)(px + q * 4);
    atomicAdd(pc + q * 4 + 0, x4.x);
    atomicAdd(pc + q * 4 + 1, x4.y);
    atomicAdd(pc + q * 4 + 2, x4.z);
    atomicAdd(pc + q * 4 + 3, x4.w);
  }
  if (lane == 0) atomicAdd(&cnt[s], 1.0f);
}

__global__ __launch_bounds__(256) void k_ctx_div(
    float* __restrict__ ctx, const float* __restrict__ cnt, int total)
{
  int i = blockIdx.x * 256 + threadIdx.x;
  if (i < total) ctx[i] /= fmaxf(cnt[i >> 8], 1.0f);
}

__global__ __launch_bounds__(256) void k_concat(
    const float* __restrict__ x, const float* __restrict__ xloc,
    float* __restrict__ xc, int rows)
{
  int i = blockIdx.x * 256 + threadIdx.x;
  if (i >= rows * XCD) return;
  int n = i / XCD, c = i - n * XCD;
  xc[i] = (c < 256) ? x[(size_t)((unsigned)n * 256u + (unsigned)c)]
                    : xloc[(size_t)((unsigned)n * 64u + (unsigned)(c - 256))];
}

__global__ __launch_bounds__(256) void k_edge_dot(
    const float* __restrict__ t, const float* __restrict__ w2,
    const float* __restrict__ b2, float* __restrict__ outp, int E)
{
  int e = blockIdx.x * 8 + (threadIdx.x >> 5);
  if (e >= E) return;
  int lane = threadIdx.x & 31;
  int c0 = lane * 8;
  const float* pt = t + (size_t)((unsigned)e * 256u + (unsigned)c0);
  float s = 0.0f;
#pragma unroll
  for (int q = 0; q < 2; ++q) {
    float4 t4 = *(const float4*)(pt + q * 4);
    float4 w4 = *(const float4*)(w2 + c0 + q * 4);
    s += t4.x * w4.x + t4.y * w4.y + t4.z * w4.z + t4.w * w4.w;
  }
#pragma unroll
  for (int o = 16; o > 0; o >>= 1) s += __shfl_xor(s, o, 32);
  if (lane == 0) outp[e] = s + b2[0];
}

// fp32 [K,Nn] -> bf16 transposed [Nn,K], tiled via LDS (both dims %32 == 0)
__global__ __launch_bounds__(256) void k_cvt_t(
    const float* __restrict__ W, unsigned short* __restrict__ WT, int K, int Nn)
{
  __shared__ unsigned short tile[32][33];
  int k0 = blockIdx.x * 32, n0 = blockIdx.y * 32;
#pragma unroll
  for (int i = 0; i < 4; ++i) {
    int k = threadIdx.y + i * 8;
    tile[k][threadIdx.x] =
        f32_bf16_rne(W[(size_t)((unsigned)(k0 + k) * (unsigned)Nn + (unsigned)(n0 + threadIdx.x))]);
  }
  __syncthreads();
#pragma unroll
  for (int i = 0; i < 4; ++i) {
    int n = threadIdx.y + i * 8;
    WT[(size_t)((unsigned)(n0 + n) * (unsigned)K + (unsigned)(k0 + threadIdx.x))] =
        tile[threadIdx.x][n];
  }
}

__global__ __launch_bounds__(256) void k_fill_f32(float* p, float v, int n) {
  int i = blockIdx.x * 256 + threadIdx.x; if (i < n) p[i] = v;
}
__global__ __launch_bounds__(256) void k_fill_u32(unsigned* p, unsigned v, int n) {
  int i = blockIdx.x * 256 + threadIdx.x; if (i < n) p[i] = v;
}

// ----------------------------- host side -----------------------------------
static inline int cdiv(int a, int b) { return (a + b - 1) / b; }

static void gemm(hipStream_t st, const float* A, const unsigned short* WT,
                 const float* bias, float* C, int M, int K, int Nn, int relu) {
  dim3 grid(cdiv(M, 64), Nn / 64);
  k_gemm<<<grid, 128, 0, st>>>(A, WT, bias, C, M, K, Nn, relu);
}

extern "C" void kernel_launch(void* const* d_in, const int* in_sizes, int n_in,
                              void* d_out, int out_size, void* d_ws, size_t ws_size,
                              hipStream_t stream) {
  const float* x0  = (const float*)d_in[0];
  const int*   src = (const int*)d_in[1];
  const int*   dst = src + NE;
  auto F = [&](int slot) { return (const float*)d_in[slot]; };

  // ---- workspace bump allocator ----
  size_t off = 0;
  auto bump = [&](size_t bytes) -> void* {
    void* p = (char*)d_ws + off;
    off += (bytes + 255) & ~(size_t)255;
    return p;
  };
  unsigned short* wb_gat[3][3];
  for (int i = 0; i < 3; ++i) {
    size_t k = (i == 0) ? DIN : HDIM;
    for (int j = 0; j < 3; ++j) wb_gat[i][j] = (unsigned short*)bump(k * HDIM * 2);
  }
  unsigned short* wb_loc1 = (unsigned short*)bump((size_t)HDIM * LCDIM * 2);
  unsigned short* wb_loc2 = (unsigned short*)bump((size_t)LCDIM * LCDIM * 2);
  unsigned short* wb_ei1  = (unsigned short*)bump((size_t)EIK * EHDIM * 2);
  unsigned short* wb_ei2  = (unsigned short*)bump((size_t)EHDIM * EHDIM * 2);
  unsigned short* wb_upd[3][2];
  for (int j = 0; j < 3; ++j)
    for (int m = 0; m < 2; ++m) wb_upd[j][m] = (unsigned short*)bump((size_t)EHDIM * EHDIM * 2);
  unsigned short* wb_eo1 = (unsigned short*)bump((size_t)EHDIM * EHDIM * 2);

  float* xl     = (float*)bump((size_t)NN * HDIM * 4);
  float* xr     = (float*)bump((size_t)NN * HDIM * 4);
  float* res    = (float*)bump((size_t)NN * HDIM * 4);
  float* attout = (float*)bump((size_t)NN * HDIM * 4);
  float* xbufA  = (float*)bump((size_t)NN * HDIM * 4);
  float* xbufB  = (float*)bump((size_t)NN * HDIM * 4);
  unsigned* lmax = (unsigned*)bump((size_t)NN * 4 * 4);
  float* den    = (float*)bump((size_t)NN * 4 * 4);
  float* exb    = (float*)bump((size_t)(NE + NN) * 4 * 4);
  float* ctx    = (float*)bump((size_t)NN * HDIM * 4);
  float* cnt    = (float*)bump((size_t)NN * 4);
  float* tl1    = (float*)bump((size_t)NN * LCDIM * 4);
  float* xloc   = (float*)bump((size_t)NN * LCDIM * 4);
  float* xc     = (float*)bump((size_t)NN * XCD * 4);
  float* t1     = (float*)bump((size_t)NE * EHDIM * 4);
  float* t2     = (float*)bump((size_t)NE * EHDIM * 4);
  float* ef     = (float*)bump((size_t)NE * EHDIM * 4);
  if (off > ws_size) return;  // workspace too small; bail without OOB writes

  // ---- weight conversion fp32 -> bf16 transposed [Nn,K], every call ----
  auto convT = [&](int slot, unsigned short* dp, int K, int Nn) {
    dim3 grid(K / 32, Nn / 32), blk(32, 8);
    k_cvt_t<<<grid, blk, 0, stream>>>(F(slot), dp, K, Nn);
  };
  for (int i = 0; i < 3; ++i) {
    int base = 2 + i * 9;
    int k = (i == 0) ? DIN : HDIM;
    convT(base + 0, wb_gat[i][0], k, HDIM);   // lin_l.W
    convT(base + 2, wb_gat[i][1], k, HDIM);   // lin_r.W
    convT(base + 5, wb_gat[i][2], k, HDIM);   // Wres
  }
  convT(29, wb_loc1, HDIM, LCDIM);
  convT(31, wb_loc2, LCDIM, LCDIM);
  convT(33, wb_ei1,  EIK, EHDIM);
  convT(35, wb_ei2,  EHDIM, EHDIM);
  for (int j = 0; j < 3; ++j) {
    convT(37 + j * 8 + 0, wb_upd[j][0], EHDIM, EHDIM);
    convT(37 + j * 8 + 4, wb_upd[j][1], EHDIM, EHDIM);
  }
  convT(61, wb_eo1, EHDIM, EHDIM);

  const int EN = NE + NN;

  // ---- 3 GATv2 layers ----
  const float* xcur = x0;
  int Kin = DIN;
  for (int i = 0; i < 3; ++i) {
    int base = 2 + i * 9;
    gemm(stream, xcur, wb_gat[i][0], F(base + 1), xl,  NN, Kin, HDIM, 0);
    gemm(stream, xcur, wb_gat[i][1], F(base + 3), xr,  NN, Kin, HDIM, 0);
    gemm(stream, xcur, wb_gat[i][2], nullptr,     res, NN, Kin, HDIM, 0);
    k_fill_u32<<<cdiv(NN * 4, 256), 256, 0, stream>>>(lmax, 0u, NN * 4);
    k_fill_f32<<<cdiv(NN * 4, 256), 256, 0, stream>>>(den, 0.0f, NN * 4);
    k_fill_f32<<<cdiv(NN * HDIM, 256), 256, 0, stream>>>(attout, 0.0f, NN * HDIM);
    k_att_logit<<<cdiv(EN, 8), 256, 0, stream>>>(xl, xr, src, dst, F(base + 4), exb, lmax, NE, NN);
    k_att_ex<<<cdiv(EN * 4, 256), 256, 0, stream>>>(exb, lmax, den, dst, NE, NN);
    k_att_scatter<<<cdiv(EN, 8), 256, 0, stream>>>(xl, exb, den, src, dst, attout, NE, NN);
    float* xo = (i & 1) ? xbufB : xbufA;
    k_layernorm<<<cdiv(NN, 8), 256, 0, stream>>>(attout, res, F(base + 6), F(base + 7),
                                                 F(base + 8), nullptr, xo, NN, 1);
    xcur = xo; Kin = HDIM;
  }

  // ---- LocalContextGather + local MLP + concat ----
  k_fill_f32<<<cdiv(NN * HDIM, 256), 256, 0, stream>>>(ctx, 0.0f, NN * HDIM);
  k_fill_f32<<<cdiv(NN, 256), 256, 0, stream>>>(cnt, 0.0f, NN);
  k_ctx_scatter<<<cdiv(NE, 8), 256, 0, stream>>>(xcur, src, dst, ctx, cnt, NE);
  k_ctx_div<<<cdiv(NN * HDIM, 256), 256, 0, stream>>>(ctx, cnt, NN * HDIM);
  gemm(stream, ctx, wb_loc1, F(30), tl1,  NN, HDIM,  LCDIM, 1);
  gemm(stream, tl1, wb_loc2, F(32), xloc, NN, LCDIM, LCDIM, 0);
  k_concat<<<cdiv(NN * XCD, 256), 256, 0, stream>>>(xcur, xloc, xc, NN);

  // ---- edge_init (fused 1280-wide feature build) ----
  {
    dim3 grid(cdiv(NE, 64), EHDIM / 64);
    k_gemm_edgefeat<<<grid, 128, 0, stream>>>(xc, src, dst, wb_ei1, F(34), t1, NE, 1);
  }
  gemm(stream, t1, wb_ei2, F(36), ef, NE, EHDIM, EHDIM, 0);

  // ---- 3 residual update blocks ----
  for (int j = 0; j < 3; ++j) {
    int base = 37 + j * 8;
    gemm(stream, ef, wb_upd[j][0], F(base + 1), t1, NE, EHDIM, EHDIM, 1);
    k_layernorm<<<cdiv(NE, 8), 256, 0, stream>>>(t1, nullptr, nullptr, F(base + 2),
                                                 F(base + 3), nullptr, t1, NE, 0);
    gemm(stream, t1, wb_upd[j][1], F(base + 5), t2, NE, EHDIM, EHDIM, 0);
    k_layernorm<<<cdiv(NE, 8), 256, 0, stream>>>(t2, nullptr, nullptr, F(base + 6),
                                                 F(base + 7), ef, ef, NE, 0);
  }

  // ---- output head ----
  gemm(stream, ef, wb_eo1, F(62), t1, NE, EHDIM, EHDIM, 1);
  k_edge_dot<<<cdiv(NE, 8), 256, 0, stream>>>(t1, F(63), F(64), (float*)d_out, NE);
}
